// SentenceClassifier_85693187490154
// MI455X (gfx1250) — compile-verified, matrix operand order checked by reading, not run on hardware
//
#include <hip/hip_runtime.h>
#include <hip/hip_bf16.h>

// Problem constants (match reference)
#define Bn   64
#define Tn   512
#define En   512
#define Hn   1024
#define Cn   32
#define KTOT (En + Hn)   // 1536
#define G4H  (4 * Hn)    // 4096
#define NKT  (KTOT / 32) // 48 K-tiles of 32
#define NKTX (En / 32)   // 16 K-tiles covering x portion

typedef __attribute__((ext_vector_type(16))) __bf16 v16bf;
typedef __attribute__((ext_vector_type(8)))  float  v8f;

union BF16Tile {
  v16bf v;
  unsigned short u[16];
  uint4 q[2];
};

__device__ __forceinline__ unsigned short f32_to_bf16_rne(float f) {
  unsigned u = __float_as_uint(f);
  u += 0x7FFFu + ((u >> 16) & 1u);   // round-to-nearest-even
  return (unsigned short)(u >> 16);
}

__device__ __forceinline__ float sigmoidf_fast(float v) {
  return 1.0f / (1.0f + __expf(-v));
}

// ---- one-time prep: W [K, 4H] f32 -> Wt [4H, K] bf16 (transposed: contiguous-in-K B loads)
__global__ void lstm_prep_wt(const float* __restrict__ W, unsigned short* __restrict__ Wt) {
  int idx = blockIdx.x * blockDim.x + threadIdx.x;
  if (idx >= G4H * KTOT) return;
  int col = idx / KTOT;
  int k   = idx - col * KTOT;
  Wt[idx] = f32_to_bf16_rne(W[(size_t)k * G4H + col]);
}

// ---- one-time prep: x f32 -> bf16 (same [B,T,E] layout)
__global__ void lstm_prep_x(const float* __restrict__ x, unsigned short* __restrict__ xb,
                            int n) {
  int idx = blockIdx.x * blockDim.x + threadIdx.x;
  if (idx < n) xb[idx] = f32_to_bf16_rne(x[idx]);
}

__global__ void lstm_zero_u32(unsigned* __restrict__ p, int n) {
  int idx = blockIdx.x * blockDim.x + threadIdx.x;
  if (idx < n) p[idx] = 0u;
}

// ---- one LSTM timestep: gates = [x_t, h] @ W + b, then fused state update.
// One wave handles ALL 64 batch rows (4 row-tiles) for one 16-column block of H:
// each weight tile is loaded once and feeds 4 WMMAs. grid = 64 blocks x wave32.
__global__ void __launch_bounds__(32)
lstm_step_wmma(const unsigned short* __restrict__ xb,   // [B, T, E] bf16
               const unsigned short* __restrict__ Wt,   // [4H, KTOT] bf16
               const float* __restrict__ bias,          // [4H]
               float* __restrict__ cst,                 // [B, H] cell state (in-place)
               const unsigned short* __restrict__ hbin, // [B, H] bf16 h_{t-1}
               unsigned short* __restrict__ hbout,      // [B, H] bf16 h_t
               float* __restrict__ hf32,                // [B, H] f32 h_t (for classifier)
               int t)
{
  const int lane = threadIdx.x;          // 0..31
  const int cb   = blockIdx.x;           // col block over H (0..63)
  const int kb   = (lane >> 4) * 8;      // K-base per half-wave: 0 or 8
  const int Mloc = lane & 15;
  const int colH = cb * 16 + Mloc;

  // accumulators [row-block][gate], bias pre-loaded (same column for all row-blocks)
  v8f acc[4][4];
#pragma unroll
  for (int g = 0; g < 4; ++g) {
    float bg = bias[g * Hn + colH];
#pragma unroll
    for (int rb = 0; rb < 4; ++rb) {
#pragma unroll
      for (int r = 0; r < 8; ++r) acc[rb][g][r] = bg;
    }
  }

  // A-source pointers per row-block (already offset by half-wave K-base)
  const unsigned short* pax[4];
  const unsigned short* pah[4];
#pragma unroll
  for (int rb = 0; rb < 4; ++rb) {
    int rowA = rb * 16 + Mloc;
    pax[rb] = xb   + (size_t)rowA * Tn * En + (size_t)t * En + kb;
    pah[rb] = hbin + (size_t)rowA * Hn + kb;
  }
  // B base pointers, one per gate
  const unsigned short* pb[4];
#pragma unroll
  for (int g = 0; g < 4; ++g) pb[g] = Wt + (size_t)(g * Hn + colH) * KTOT + kb;

#define DO_KTILE(PAS, AOFF, KOFF)                                          \
  {                                                                        \
    BF16Tile A[4];                                                         \
    _Pragma("unroll")                                                      \
    for (int rb = 0; rb < 4; ++rb) {                                       \
      A[rb].q[0] = *(const uint4*)((PAS)[rb] + (AOFF) + 0);                \
      A[rb].q[1] = *(const uint4*)((PAS)[rb] + (AOFF) + 16);               \
    }                                                                      \
    _Pragma("unroll")                                                      \
    for (int g = 0; g < 4; ++g) {                                          \
      BF16Tile Bt;                                                         \
      Bt.q[0] = *(const uint4*)(pb[g] + (KOFF) + 0);                       \
      Bt.q[1] = *(const uint4*)(pb[g] + (KOFF) + 16);                      \
      __builtin_prefetch(pb[g] + (KOFF) + 32, 0, 1);                       \
      _Pragma("unroll")                                                    \
      for (int rb = 0; rb < 4; ++rb) {                                     \
        acc[rb][g] = __builtin_amdgcn_wmma_f32_16x16x32_bf16(              \
            false, A[rb].v, false, Bt.v, (short)0, acc[rb][g],             \
            false, false);                                                 \
      }                                                                    \
    }                                                                      \
  }

  // x-region K tiles: K = 0 .. E
  for (int kt = 0; kt < NKTX; ++kt) {
    DO_KTILE(pax, kt * 32, kt * 32)
  }
  // h-region K tiles: K = E .. E+H
  for (int kt = NKTX; kt < NKT; ++kt) {
    DO_KTILE(pah, (kt - NKTX) * 32, kt * 32)
  }
#undef DO_KTILE

  // ---- fused gate nonlinearity + cell/hidden update
  // C/D layout: VGPR r, lanes 0-15 -> M=r, lanes 16-31 -> M=r+8; N = lane&15
#pragma unroll
  for (int rb = 0; rb < 4; ++rb) {
#pragma unroll
    for (int r = 0; r < 8; ++r) {
      int rowM = rb * 16 + r + ((lane >> 4) << 3);
      size_t idx = (size_t)rowM * Hn + colH;
      float ig = acc[rb][0][r], jg = acc[rb][1][r];
      float fg = acc[rb][2][r], og = acc[rb][3][r];
      float c_old = cst[idx];
      float nc = c_old * sigmoidf_fast(fg + 1.0f) + sigmoidf_fast(ig) * tanhf(jg);
      float nh = tanhf(nc) * sigmoidf_fast(og);
      cst[idx]   = nc;
      hf32[idx]  = nh;
      hbout[idx] = f32_to_bf16_rne(nh);
    }
  }
}

// ---- classifier head: pred = h_last @ out_W + out_b  (tiny: 4 MFLOP)
__global__ void classifier_logits(const float* __restrict__ h,
                                  const float* __restrict__ oW,
                                  const float* __restrict__ ob,
                                  float* __restrict__ out) {
  int idx = blockIdx.x * blockDim.x + threadIdx.x;
  if (idx >= Bn * Cn) return;
  int b = idx / Cn, c = idx - (idx / Cn) * Cn;
  const float* hr = h + (size_t)b * Hn;
  float s = ob[c];
#pragma unroll 4
  for (int k = 0; k < Hn; ++k) s = fmaf(hr[k], oW[k * Cn + c], s);
  out[idx] = s;
}

// ---- per-row softmax + CE term; one wave per row, C==32 -> one col per lane
__global__ void __launch_bounds__(32)
softmax_ce(const float* __restrict__ pred, const float* __restrict__ y,
           float* __restrict__ outsm, float* __restrict__ ce) {
  int r = blockIdx.x;
  int lane = threadIdx.x;
  float p = pred[r * Cn + lane];
  float m = p;
  for (int off = 16; off > 0; off >>= 1) m = fmaxf(m, __shfl_xor(m, off, 32));
  float ex = __expf(p - m);
  float s = ex;
  for (int off = 16; off > 0; off >>= 1) s += __shfl_xor(s, off, 32);
  outsm[r * Cn + lane] = ex / s;
  float logp = (p - m) - __logf(s);
  float tterm = y[r * Cn + lane] * logp;
  for (int off = 16; off > 0; off >>= 1) tterm += __shfl_xor(tterm, off, 32);
  if (lane == 0) ce[r] = -tterm;
}

// ---- fixed-order mean over 64 rows (deterministic, no FP atomics)
__global__ void __launch_bounds__(32)
ce_mean(const float* __restrict__ ce, float* __restrict__ cost) {
  int lane = threadIdx.x;
  float v = ce[lane] + ce[lane + 32];
  for (int off = 16; off > 0; off >>= 1) v += __shfl_xor(v, off, 32);
  if (lane == 0) cost[0] = v * (1.0f / Bn);
}

extern "C" void kernel_launch(void* const* d_in, const int* in_sizes, int n_in,
                              void* d_out, int out_size, void* d_ws, size_t ws_size,
                              hipStream_t stream) {
  (void)in_sizes; (void)n_in; (void)out_size; (void)ws_size;
  const float* x    = (const float*)d_in[0];
  const float* y    = (const float*)d_in[1];
  // d_in[2] = seqlen (unused by the reference's dynamic_rnn path)
  const float* W    = (const float*)d_in[3];
  const float* bias = (const float*)d_in[4];
  const float* oW   = (const float*)d_in[5];
  const float* ob   = (const float*)d_in[6];
  float* out = (float*)d_out;

  // Workspace layout (byte offsets; all 256B-aligned)
  char* ws = (char*)d_ws;
  float*          cst  = (float*)ws;                                   // 256 KB
  unsigned short* hb0  = (unsigned short*)(ws + 262144);               // 128 KB (zero region contiguous with cst)
  unsigned short* hb1  = (unsigned short*)(ws + 262144 + 131072);      // 128 KB
  float*          hf32 = (float*)(ws + 524288);                        // 256 KB
  float*          ce   = (float*)(ws + 786432);                        // 256 B
  unsigned short* Wt   = (unsigned short*)(ws + 790528);               // 12.58 MB
  unsigned short* xb   = (unsigned short*)(ws + 790528 + (size_t)G4H * KTOT * 2); // 33.5 MB

  {
    int total = G4H * KTOT;
    lstm_prep_wt<<<(total + 255) / 256, 256, 0, stream>>>(W, Wt);
  }
  {
    int total = Bn * Tn * En;
    lstm_prep_x<<<(total + 255) / 256, 256, 0, stream>>>(x, xb, total);
  }
  {
    // zero c (f32) and hb0 (bf16): contiguous 393216 bytes = 98304 u32
    int total = (Bn * Hn * 4 + Bn * Hn * 2) / 4;
    lstm_zero_u32<<<(total + 255) / 256, 256, 0, stream>>>((unsigned*)cst, total);
  }

  for (int t = 0; t < Tn; ++t) {
    const unsigned short* hbin = (t & 1) ? hb1 : hb0;
    unsigned short*      hbout = (t & 1) ? hb0 : hb1;
    lstm_step_wmma<<<dim3(64), 32, 0, stream>>>(xb, Wt, bias, cst, hbin, hbout, hf32, t);
  }

  classifier_logits<<<(Bn * Cn + 255) / 256, 256, 0, stream>>>(hf32, oW, ob, out);
  softmax_ce<<<Bn, 32, 0, stream>>>(out, y, out + Bn * Cn, ce);
  ce_mean<<<1, 32, 0, stream>>>(ce, out + 2 * Bn * Cn);
}
